// Lea_28690381537780
// MI455X (gfx1250) — compile-verified, matrix operand order checked by reading, not run on hardware
//
#include <hip/hip_runtime.h>
#include <hip/hip_bf16.h>

typedef __attribute__((ext_vector_type(16))) _Float16 v16h;
typedef __attribute__((ext_vector_type(8)))  float    v8f;

#define B_TOT 4096
#define T_LEN 301
#define IN_D  40
#define H_D   64
#define G4    256     // 4*H
#define BTILE 16      // batch rows per block (256 blocks -> fill the WGPs)
#define NTHREADS 256

// ---- LDS layout (bytes) ----
// wih  : swizzled f16 B-fragments, 16384 halves = 32768 B   @ 0
// whh  : swizzled f16 B-fragments, 16384 halves = 32768 B   @ 32768
// bias : 256 f32                    = 1024 B                @ 65536
// xbuf : 2 x [16][64] f16 (padded)  = 4096 B                @ 66560
// z    : [16][256] f32              = 16384 B               @ 70656
// h    : [16][64] f16               = 2048 B                @ 87040
#define SMEM_BYTES 89088

__device__ __forceinline__ float tanhf_hw(float x) {
#if __has_builtin(__builtin_amdgcn_tanhf)
  return __builtin_amdgcn_tanhf(x);       // v_tanh_f32: single trans op
#else
  x = fminf(15.0f, fmaxf(-15.0f, x));
  float e = __expf(2.0f * x);
  return (e - 1.0f) / (e + 1.0f);
#endif
}
__device__ __forceinline__ float sigmf(float x) {
  return 0.5f + 0.5f * tanhf_hw(0.5f * x); // 1 trans + 2 VALU
}

// 16x32 f16 A-fragment from LDS, row-major [m][k] with row stride ldk halves.
// ISA layout: lanes 0-15 -> M=lane, K in {kb..kb+7, kb+16..kb+23} with kb=0;
//             lanes 16-31 -> M=lane-16, kb=8.
__device__ __forceinline__ v16h load_frag_A(const _Float16* base, int k0,
                                            int ldk, int lane) {
  int m  = lane & 15;
  int kb = (lane < 16) ? 0 : 8;
  const _Float16* p = base + m * ldk + k0 + kb;
  v16h a;
#pragma unroll
  for (int j = 0; j < 8; ++j) a[j] = p[j];          // contiguous 16B -> b128
#pragma unroll
  for (int j = 0; j < 8; ++j) a[8 + j] = p[16 + j]; // contiguous 16B -> b128
  return a;
}

__global__ void __launch_bounds__(NTHREADS)
lstm_fused_wmma(const float* __restrict__ xg,   const float* __restrict__ wihg,
                const float* __restrict__ whhg, const float* __restrict__ bihg,
                const float* __restrict__ bhhg, const float* __restrict__ wclf,
                const float* __restrict__ bclf, float* __restrict__ out) {
  extern __shared__ __align__(16) char smem[];
  _Float16* wih  = (_Float16*)(smem);
  _Float16* whh  = (_Float16*)(smem + 32768);
  float*    bias = (float*)   (smem + 65536);
  _Float16* xbuf = (_Float16*)(smem + 66560);
  float*    z    = (float*)   (smem + 70656);
  _Float16* hl   = (_Float16*)(smem + 87040);

  const int tid  = threadIdx.x;
  const int lane = tid & 31;
  const int wave = tid >> 5;
  const int b0   = blockIdx.x * BTILE;

  // ---- init: swizzle weights into WMMA B-fragment order ----
  // fragment f = (sn*2 + kc)*32 + lane holds halves j=0..15:
  //   n = sn*16 + (lane&15); kb = lane<16?0:8; k = kc*32 + kb + (j<8 ? j : j+8)
  for (int idx = tid; idx < 16384; idx += NTHREADS) {
    int j    = idx & 15;
    int ln   = (idx >> 4) & 31;
    int kc   = (idx >> 9) & 1;
    int sn   = idx >> 10;
    int n    = sn * 16 + (ln & 15);
    int kb   = (ln < 16) ? 0 : 8;
    int k    = kc * 32 + kb + ((j < 8) ? j : (j + 8));
    wih[idx] = (_Float16)((k < IN_D) ? wihg[n * IN_D + k] : 0.0f);
    whh[idx] = (_Float16)(whhg[n * H_D + k]);
  }
  bias[tid] = bihg[tid] + bhhg[tid];
  for (int idx = tid; idx < BTILE * H_D; idx += NTHREADS) hl[idx] = (_Float16)0.0f;
  for (int idx = tid; idx < 2 * BTILE * 64; idx += NTHREADS) xbuf[idx] = (_Float16)0.0f;
  // x for t=0 into buffer 0 (16*40 = 640 elements, <=3 per thread)
#pragma unroll
  for (int q = 0; q < 3; ++q) {
    int e = tid + q * NTHREADS;
    if (e < BTILE * IN_D) {
      int m = e / IN_D, k = e % IN_D;
      xbuf[m * 64 + k] = (_Float16)xg[((size_t)(b0 + m) * T_LEN + 0) * IN_D + k];
    }
  }
  __syncthreads();

  // ---- per-thread recurrent state for elementwise phase ----
  const int em = tid >> 4;        // batch row within tile (0..15)
  const int jb = (tid & 15) * 4;  // hidden-unit base (4 units per thread)
  float cpriv[4];
#pragma unroll
  for (int jj = 0; jj < 4; ++jj) cpriv[jj] = 0.0f;
  float accClf = 0.0f;

  // each wave owns N cols [wave*32, wave*32+32): subtiles sn = wave*2, wave*2+1
  const v16h* wihf = (const v16h*)wih;
  const v16h* whhf = (const v16h*)whh;

  for (int t = 0; t < T_LEN; ++t) {
    // -- phase A: issue global loads of x(t+1); latency hidden under WMMA --
    float xr[3];
    if (t + 1 < T_LEN) {
#pragma unroll
      for (int q = 0; q < 3; ++q) {
        int e = tid + q * NTHREADS;
        if (e < BTILE * IN_D) {
          int m = e / IN_D, k = e % IN_D;
          xr[q] = xg[((size_t)(b0 + m) * T_LEN + (t + 1)) * IN_D + k];
        }
      }
    }
    if (t + 2 < T_LEN)
      __builtin_prefetch(&xg[((size_t)(b0 + em) * T_LEN + (t + 2)) * IN_D], 0, 0);

    // -- phase B: z = x_t @ w_ih^T + h @ w_hh^T + bias via WMMA --
    const _Float16* xb = xbuf + (t & 1) * (BTILE * 64);
    v16h ah0 = load_frag_A(hl, 0, 64, lane);
    v16h ah1 = load_frag_A(hl, 32, 64, lane);
    v16h ax0 = load_frag_A(xb, 0, 64, lane);
    v16h ax1 = load_frag_A(xb, 32, 64, lane);

#pragma unroll
    for (int s = 0; s < 2; ++s) {
      int sn = wave * 2 + s;            // global 16-col subtile index
      v16h bi0 = wihf[(sn * 2 + 0) * 32 + lane];
      v16h bi1 = wihf[(sn * 2 + 1) * 32 + lane];
      v16h bh0 = whhf[(sn * 2 + 0) * 32 + lane];
      v16h bh1 = whhf[(sn * 2 + 1) * 32 + lane];
      // two independent accumulation chains -> half the WMMA RAW latency
      v8f acc0 = {};
      v8f acc1 = {};
      acc0 = __builtin_amdgcn_wmma_f32_16x16x32_f16(false, ax0, false, bi0,
                                                    (short)0, acc0, false, false);
      acc1 = __builtin_amdgcn_wmma_f32_16x16x32_f16(false, ax1, false, bi1,
                                                    (short)0, acc1, false, false);
      acc0 = __builtin_amdgcn_wmma_f32_16x16x32_f16(false, ah0, false, bh0,
                                                    (short)0, acc0, false, false);
      acc1 = __builtin_amdgcn_wmma_f32_16x16x32_f16(false, ah1, false, bh1,
                                                    (short)0, acc1, false, false);
      // C/D layout: VGPR r -> M = r + (lane<16?0:8), N = sn*16 + (lane&15)
      int n    = sn * 16 + (lane & 15);
      int mofs = (lane < 16) ? 0 : 8;
      float bv = bias[n];
#pragma unroll
      for (int r = 0; r < 8; ++r)
        z[(mofs + r) * G4 + n] = acc0[r] + acc1[r] + bv;
    }
    __syncthreads();

    // -- phase C: gates, state update, fused classifier accumulation --
    {
      const float* wc = wclf + (size_t)t * H_D + jb;
#pragma unroll
      for (int jj = 0; jj < 4; ++jj) {
        int j    = jb + jj;
        float zi = z[em * G4 + j];
        float zf = z[em * G4 + 64 + j];
        float zg = z[em * G4 + 128 + j];
        float zo = z[em * G4 + 192 + j];
        float cc = sigmf(zf) * cpriv[jj] + sigmf(zi) * tanhf_hw(zg);
        cpriv[jj] = cc;
        float hh = sigmf(zo) * tanhf_hw(cc);
        hl[em * H_D + j] = (_Float16)hh;
        accClf += hh * wc[jj];
      }
    }

    // -- phase D: commit x(t+1) into the alternate LDS buffer --
    if (t + 1 < T_LEN) {
      _Float16* xb2 = xbuf + ((t + 1) & 1) * (BTILE * 64);
#pragma unroll
      for (int q = 0; q < 3; ++q) {
        int e = tid + q * NTHREADS;
        if (e < BTILE * IN_D) {
          int m = e / IN_D, k = e % IN_D;
          xb2[m * 64 + k] = (_Float16)xr[q];
        }
      }
    }
    __syncthreads();
  }

  // ---- classifier reduction: 16 partials per batch row (reuse z as scratch) ----
  float* red = z;
  red[tid] = accClf;
  __syncthreads();
  if (tid < BTILE) {
    float s = 0.0f;
#pragma unroll
    for (int q = 0; q < 16; ++q) s += red[tid * 16 + q];
    out[b0 + tid] = s + bclf[0];
  }
}

extern "C" void kernel_launch(void* const* d_in, const int* in_sizes, int n_in,
                              void* d_out, int out_size, void* d_ws, size_t ws_size,
                              hipStream_t stream) {
  const float* x    = (const float*)d_in[0];
  const float* wih  = (const float*)d_in[1];
  const float* whh  = (const float*)d_in[2];
  const float* bih  = (const float*)d_in[3];
  const float* bhh  = (const float*)d_in[4];
  const float* wclf = (const float*)d_in[5];
  const float* bclf = (const float*)d_in[6];
  float* out = (float*)d_out;

  hipFuncSetAttribute(reinterpret_cast<const void*>(lstm_fused_wmma),
                      hipFuncAttributeMaxDynamicSharedMemorySize, SMEM_BYTES);
  lstm_fused_wmma<<<dim3(B_TOT / BTILE), dim3(NTHREADS), SMEM_BYTES, stream>>>(
      x, wih, whh, bih, bhh, wclf, bclf, out);
}
